// VectorQuantizedVAE_52243982188940
// MI455X (gfx1250) — matrix operand order repaired and checked by reading, not compiled, not run on hardware
//
#include <hip/hip_runtime.h>

// ---------- types for CDNA5 WMMA (wave32) ----------
typedef __attribute__((ext_vector_type(16))) __bf16        v16bf;
typedef __attribute__((ext_vector_type(8)))  float          v8f;
typedef __attribute__((ext_vector_type(8)))  unsigned short u16x8;
typedef __attribute__((ext_vector_type(4)))  unsigned int   v4u;
typedef __attribute__((ext_vector_type(8)))  int            v8i;
typedef __attribute__((ext_vector_type(4)))  int            v4i;
union frag16 { v16bf v; u16x8 h[2]; };

__device__ __forceinline__ unsigned short f2bf(float f) {
  union { float f; unsigned u; } c; c.f = f;
  unsigned r = c.u + 0x7FFFu + ((c.u >> 16) & 1u);   // round-to-nearest-even
  return (unsigned short)(r >> 16);
}

// ---------- Tensor Data Mover: 2D tile (bf16) global -> LDS ----------
// D# per CDNA5 ISA ch.8: group0 {count, lds_addr, global_addr, type=2},
// group1 {data_size=2B, tensor_dim0=tile_w, tensor_dim1=rows,
//         tile_dim0=tile_w, tile_dim1=rows, tensor_dim0_stride=stride}.
// Groups 2/3 zero (2D tensor). Issued by one wave; EXEC is ignored by TDM.
#if __has_include(<hip/amd_detail/amd_gfx1250_TDM.h>)
#define TDM_ISSUE(g0, g1) do { v4i z4 = {}; v8i z8 = {};                     \
    __builtin_amdgcn_tensor_load_to_lds(g0, g1, z4, z4, z8, 0); } while (0)
#else
#define TDM_ISSUE(g0, g1) do { v4i z4 = {};                                  \
    __builtin_amdgcn_tensor_load_to_lds(g0, g1, z4, z4, 0); } while (0)
#endif

__device__ __forceinline__ void tdm_load_2d_bf16(unsigned lds_addr, const void* gptr,
                                                 int rows, int tile_w, int stride_elems) {
  unsigned long long ga = (unsigned long long)gptr;
  v4u g0;
  g0.x = 1u;                                                    // count=1, user D#
  g0.y = lds_addr;                                              // LDS byte address
  g0.z = (unsigned)(ga & 0xffffffffu);                          // global_addr[31:0]
  g0.w = (unsigned)((ga >> 32) & 0x1ffffffu) | (2u << 30);      // addr[56:32] | type=2
  v8i g1;
  g1[0] = (int)(1u << 16);                                      // data_size=1 (2 bytes)
  g1[1] = (int)(((unsigned)tile_w & 0xffffu) << 16);            // tensor_dim0[15:0]
  g1[2] = (int)((((unsigned)tile_w >> 16) & 0xffffu) |
                (((unsigned)rows & 0xffffu) << 16));            // dim0 hi | tensor_dim1 lo
  g1[3] = (int)(((unsigned)tile_w & 0xffffu) << 16);            // dim1 hi=0 | tile_dim0
  g1[4] = (int)((unsigned)rows & 0xffffu);                      // tile_dim1 | tile_dim2=0
  g1[5] = (int)(unsigned)stride_elems;                          // tensor_dim0_stride lo32
  g1[6] = 0;                                                    // stride hi | dim1_stride lo
  g1[7] = 0;
  TDM_ISSUE(g0, g1);
}

// =====================================================================
// Weight pack: OIHW fp32 -> bf16 [Mpad][Kpad], zero-filled padding
// =====================================================================
__global__ void k_pack(const float* __restrict__ W, unsigned short* __restrict__ out,
                       int Co, int K, int Kpad, long long total) {
  long long o = (long long)blockIdx.x * blockDim.x + threadIdx.x;
  if (o >= total) return;
  int m = (int)(o / Kpad);
  int k = (int)(o - (long long)m * Kpad);
  unsigned short v = 0;
  if (m < Co && k < K) v = f2bf(W[(long long)m * K + k]);
  out[o] = v;
}

__global__ void k_emb_norm(const float* __restrict__ emb, float* __restrict__ en,
                           int K, int D) {
  int c = blockIdx.x * blockDim.x + threadIdx.x;
  if (c >= K) return;
  float s = 0.f;
  for (int d = 0; d < D; ++d) { float v = emb[c * D + d]; s += v * v; }
  en[c] = s;
}

// =====================================================================
// Generic implicit-GEMM conv (supports lhs_dilation for transpose conv)
// GEMM: M = Co, N = B*Ho*Wo, K = Ci*KH*KW (padded to mult of 32)
// Block: 256 thr = 8 waves, tile 64(M) x 128(N), K-step 32.
// A tile staged by the Tensor Data Mover (async, overlapped with the
// im2col B gather), fragments per ISA 7.12.2 bf16 layouts.
// =====================================================================
#define KSTEP 32
#define MT    64
#define NT    128
#define BPAD  40   // LDS B row stride in bf16 elems (32 + 8 to dodge bank conflicts)

__global__ __launch_bounds__(256)
void k_conv_gemm(const float* __restrict__ X, const unsigned short* __restrict__ Wp,
                 const float* __restrict__ bias, float* __restrict__ Y,
                 int B, int Ci, int Hi, int Wi, int Co, int Ho, int Wo,
                 int KH, int KW, int stride, int pad, int ldil,
                 int Kpad, int relu_in) {
  __shared__ __align__(16) unsigned short sA[MT * KSTEP];
  __shared__ __align__(16) unsigned short sB[NT * BPAD];

  const int tid  = threadIdx.x;
  const int lane = tid & 31;
  const int wave = tid >> 5;
  const int wm   = wave >> 2;          // 0..1  (M sub-tile)
  const int wn   = wave & 3;           // 0..3  (N sub-tile)
  const int n0   = blockIdx.x * NT;
  const int m0   = blockIdx.y * MT;
  const int Ntot  = B * Ho * Wo;
  const int Kreal = Ci * KH * KW;
  const int KHW   = KH * KW;
  const unsigned sA_lds = (unsigned)(size_t)(void*)sA;

  v8f acc[2][2] = {};

  for (int k0 = 0; k0 < Kpad; k0 += KSTEP) {
    __syncthreads();   // previous iteration's fragment reads complete
    // ---- stage A (weights) via TDM: 64 rows x 32 bf16, async ----
    if (wave == 0)
      tdm_load_2d_bf16(sA_lds, Wp + (size_t)m0 * Kpad + k0, MT, KSTEP, Kpad);
    // ---- stage B (im2col gather): 128 cols x 32 k, 16 elems/thread ----
    {
      int ncol = tid >> 1;
      int kk0  = (tid & 1) * 16;
      int n = n0 + ncol;
      int wo = 0, ho = 0, b = 0;
      bool okn = n < Ntot;
      if (okn) { wo = n % Wo; int t = n / Wo; ho = t % Ho; b = t / Ho; }
      #pragma unroll 4
      for (int u = 0; u < 16; ++u) {
        int k = k0 + kk0 + u;
        float v = 0.f;
        if (okn && k < Kreal) {
          int ci = k / KHW, r = k - ci * KHW;
          int kh = r / KW,  kw = r - kh * KW;
          int ihn = ho * stride - pad + kh;
          int iwn = wo * stride - pad + kw;
          if (ldil == 1) {
            if (ihn >= 0 && ihn < Hi && iwn >= 0 && iwn < Wi)
              v = X[(((size_t)b * Ci + ci) * Hi + ihn) * Wi + iwn];
          } else {
            if (ihn >= 0 && iwn >= 0 && (ihn % ldil) == 0 && (iwn % ldil) == 0) {
              int ih = ihn / ldil, iw = iwn / ldil;
              if (ih < Hi && iw < Wi)
                v = X[(((size_t)b * Ci + ci) * Hi + ih) * Wi + iw];
            }
          }
          if (relu_in) v = v > 0.f ? v : 0.f;
        }
        sB[ncol * BPAD + kk0 + u] = f2bf(v);
      }
    }
    if (wave == 0) __builtin_amdgcn_s_wait_tensorcnt(0);  // A tile landed in LDS
    __syncthreads();

    // ---- fragment loads (ISA 7.12.2 bf16 layouts) + 4 WMMAs ----
    frag16 af[2], bf2[2];
    #pragma unroll
    for (int i = 0; i < 2; ++i) {
      int m  = wm * 32 + i * 16 + (lane & 15);
      int kb = (lane < 16) ? 0 : 8;            // A: lanes 0-15 K=[0..7]+[16..23]
      af[i].h[0] = *(const u16x8*)(sA + m * KSTEP + kb);
      af[i].h[1] = *(const u16x8*)(sA + m * KSTEP + kb + 16);
    }
    #pragma unroll
    for (int j = 0; j < 2; ++j) {
      int n  = wn * 32 + j * 16 + (lane & 15);
      int kb = (lane < 16) ? 0 : 8;            // B: lanes 0-15 K=[0..7]+[16..23]
      bf2[j].h[0] = *(const u16x8*)(sB + n * BPAD + kb);
      bf2[j].h[1] = *(const u16x8*)(sB + n * BPAD + kb + 16);
    }
    #pragma unroll
    for (int i = 0; i < 2; ++i)
      #pragma unroll
      for (int j = 0; j < 2; ++j)
        acc[i][j] = __builtin_amdgcn_wmma_f32_16x16x32_bf16(
            false, af[i].v, false, bf2[j].v, (short)0, acc[i][j], false, false);
  }

  // ---- epilogue: +bias, scatter to NCHW fp32 ----
  #pragma unroll
  for (int i = 0; i < 2; ++i)
    #pragma unroll
    for (int j = 0; j < 2; ++j) {
      int col = n0 + wn * 32 + j * 16 + (lane & 15);
      if (col >= Ntot) continue;
      int wo = col % Wo; int t = col / Wo; int ho = t % Ho; int b = t / Ho;
      #pragma unroll
      for (int v = 0; v < 8; ++v) {
        int m = m0 + wm * 32 + i * 16 + v + ((lane < 16) ? 0 : 8);
        if (m >= Co) continue;
        Y[(((size_t)b * Co + m) * Ho + ho) * Wo + wo] = acc[i][j][v] + bias[m];
      }
    }
}

// =====================================================================
// BatchNorm (training-mode): per-channel stats, then normalize (+ReLU)
// =====================================================================
__global__ __launch_bounds__(256)
void k_bn_stats(const float* __restrict__ X, float* __restrict__ mean,
                float* __restrict__ istd, int B, int C, int HW) {
  __shared__ float rs[256], rs2[256];
  int c = blockIdx.x, tid = threadIdx.x;
  float s = 0.f, s2 = 0.f;
  int tot = B * HW;
  for (int i = tid; i < tot; i += 256) {
    int b = i / HW, r = i - b * HW;
    float v = X[((size_t)b * C + c) * HW + r];
    s += v; s2 += v * v;
  }
  rs[tid] = s; rs2[tid] = s2; __syncthreads();
  for (int o = 128; o > 0; o >>= 1) {
    if (tid < o) { rs[tid] += rs[tid + o]; rs2[tid] += rs2[tid + o]; }
    __syncthreads();
  }
  if (tid == 0) {
    float cnt = (float)tot;
    float m = rs[0] / cnt;
    float var = rs2[0] / cnt - m * m;
    mean[c] = m;
    istd[c] = rsqrtf(var + 1e-5f);
  }
}

__global__ void k_bn_map(const float* __restrict__ X, float* __restrict__ Y,
                         const float* __restrict__ mean, const float* __restrict__ istd,
                         const float* __restrict__ g, const float* __restrict__ b,
                         int C, int HW, int relu, long long total) {
  long long o = (long long)blockIdx.x * blockDim.x + threadIdx.x;
  if (o >= total) return;
  int c = (int)((o / HW) % C);
  float v = (X[o] - mean[c]) * istd[c] * g[c] + b[c];
  if (relu) v = v > 0.f ? v : 0.f;
  Y[o] = v;
}

// =====================================================================
// VQ: argmin_k ||z - e_k||^2 via WMMA (score = ||e||^2 - 2 z.e)
// Block = 128 thr (4 waves); each wave owns 16 pixels (z staged in LDS).
// =====================================================================
__global__ __launch_bounds__(128)
void k_vq_argmin(const float* __restrict__ Z, const unsigned short* __restrict__ Ep,
                 const float* __restrict__ en, int* __restrict__ idx,
                 int Npix, int D, int Hh, int Ww, int C, int Ktot) {
  __shared__ __align__(16) unsigned short sZ[4][16 * 264];  // 264 = 256 + 8 pad
  int lane = threadIdx.x & 31, wave = threadIdx.x >> 5;
  int p0 = blockIdx.x * 64 + wave * 16;

  for (int e = lane; e < 16 * 256; e += 32) {
    int pl = e >> 8, d = e & 255;
    int p = p0 + pl;
    float v = 0.f;
    if (p < Npix) {
      int ww = p % Ww; int t = p / Ww; int hh = t % Hh; int b = t / Hh;
      v = Z[(((size_t)b * C + d) * Hh + hh) * Ww + ww];
    }
    sZ[wave][pl * 264 + d] = f2bf(v);
  }
  __syncthreads();

  float best = 3.4e38f; int bi = 0;
  for (int ct = 0; ct < Ktot / 16; ++ct) {
    v8f acc = {};
    for (int k0 = 0; k0 < D; k0 += 32) {
      frag16 fa, fb;
      int code = ct * 16 + (lane & 15);
      int kb   = (lane < 16) ? 0 : 8;
      const unsigned short* ea = Ep + (size_t)code * D + k0 + kb;
      fa.h[0] = *(const u16x8*)ea;
      fa.h[1] = *(const u16x8*)(ea + 16);
      if (ct + 1 < Ktot / 16) __builtin_prefetch(ea + D * 16, 0, 3);  // next code tile
      int pl  = lane & 15;
      int kb2 = (lane < 16) ? 0 : 8;
      const unsigned short* eb = sZ[wave] + pl * 264 + k0 + kb2;
      fb.h[0] = *(const u16x8*)eb;
      fb.h[1] = *(const u16x8*)(eb + 16);
      acc = __builtin_amdgcn_wmma_f32_16x16x32_bf16(
          false, fa.v, false, fb.v, (short)0, acc, false, false);
    }
    #pragma unroll
    for (int v = 0; v < 8; ++v) {
      int code = ct * 16 + v + ((lane < 16) ? 0 : 8);
      float s = en[code] - 2.f * acc[v];
      if (s < best) { best = s; bi = code; }
    }
  }
  // combine the two lane-halves (same pixel column) — wave32 shuffle
  float ob = __shfl_xor(best, 16, 32);
  int   oi = __shfl_xor(bi, 16, 32);
  if (ob < best || (ob == best && oi < bi)) { best = ob; bi = oi; }
  if (lane < 16) {
    int p = p0 + lane;
    if (p < Npix) idx[p] = bi;
  }
}

__global__ void k_vq_gather(const float* __restrict__ emb, const int* __restrict__ idx,
                            float* __restrict__ zq_ws, float* __restrict__ zq_out,
                            int C, int Hh, int Ww, long long total) {
  long long o = (long long)blockIdx.x * blockDim.x + threadIdx.x;
  if (o >= total) return;
  int w = (int)(o % Ww); long long t = o / Ww;
  int h = (int)(t % Hh); t /= Hh;
  int c = (int)(t % C);  int b = (int)(t / C);
  int p = (b * Hh + h) * Ww + w;
  float v = emb[(size_t)idx[p] * C + c];
  zq_ws[o] = v; zq_out[o] = v;
}

// =====================================================================
// elementwise glue
// =====================================================================
__global__ void k_add(float* __restrict__ a, const float* __restrict__ b, long long n) {
  long long o = (long long)blockIdx.x * blockDim.x + threadIdx.x;
  if (o < n) a[o] += b[o];
}
__global__ void k_copy(float* __restrict__ d, const float* __restrict__ s, long long n) {
  long long o = (long long)blockIdx.x * blockDim.x + threadIdx.x;
  if (o < n) d[o] = s[o];
}
__global__ void k_tanh(float* __restrict__ a, long long n) {
  long long o = (long long)blockIdx.x * blockDim.x + threadIdx.x;
  if (o < n) a[o] = tanhf(a[o]);
}

// =====================================================================
// host orchestration
// =====================================================================
extern "C" void kernel_launch(void* const* d_in, const int* in_sizes, int n_in,
                              void* d_out, int out_size, void* d_ws, size_t ws_size,
                              hipStream_t stream) {
  (void)in_sizes; (void)n_in; (void)out_size; (void)ws_size;
  const float* x = (const float*)d_in[0];
  // params flattened in dict order
  const float* enc_w1 = (const float*)d_in[1];  const float* enc_b1 = (const float*)d_in[2];
  const float* bn1_g  = (const float*)d_in[3];  const float* bn1_b  = (const float*)d_in[4];
  const float* enc_w2 = (const float*)d_in[5];  const float* enc_b2 = (const float*)d_in[6];
  const float* emb    = (const float*)d_in[23];
  const float* dec_w1 = (const float*)d_in[40]; const float* dec_b1 = (const float*)d_in[41];
  const float* dbn_g  = (const float*)d_in[42]; const float* dbn_b  = (const float*)d_in[43];
  const float* dec_w2 = (const float*)d_in[44]; const float* dec_b2 = (const float*)d_in[45];
  const int rb_base[4] = {7, 15, 24, 32};   // rb1, rb2, rb3, rb4

  float* out = (float*)d_out;
  float* o_xt = out;                       // 32*3*256*128   = 3,145,728
  float* o_ze = out + 3145728;             // 32*256*64*32   = 16,777,216
  float* o_zq = o_ze + 16777216;

  // ---- workspace bump allocator ----
  char* ws = (char*)d_ws;
  size_t off = 0;
  auto alloc = [&](size_t bytes) -> void* {
    void* p = ws + off;
    off = (off + bytes + 255) & ~(size_t)255;
    return p;
  };
  float* big0 = (float*)alloc((size_t)67108864 * 4);   // 32x256x128x64
  float* midA = (float*)alloc((size_t)16777216 * 4);   // 32x256x64x32
  float* midB = (float*)alloc((size_t)16777216 * 4);
  float* midC = (float*)alloc((size_t)16777216 * 4);
  float* mean = (float*)alloc(256 * 4);
  float* istd = (float*)alloc(256 * 4);
  float* en   = (float*)alloc(512 * 4);
  int*   idx  = (int*)alloc(65536 * 4);
  auto wpack = [&](int Mpad, int Kpad) {
    return (unsigned short*)alloc((size_t)Mpad * Kpad * 2);
  };
  unsigned short* p_enc1 = wpack(256, 64);
  unsigned short* p_enc2 = wpack(256, 4096);
  unsigned short* p_w3[4]; unsigned short* p_w1[4];
  for (int r = 0; r < 4; ++r) { p_w3[r] = wpack(256, 2304); p_w1[r] = wpack(256, 256); }
  unsigned short* p_dec1 = wpack(256, 4096);
  unsigned short* p_dec2 = wpack(64, 4096);
  unsigned short* p_emb  = wpack(512, 256);

  auto pack = [&](const float* w, unsigned short* o, int Co, int K, int Mpad, int Kpad) {
    long long tot = (long long)Mpad * Kpad;
    k_pack<<<dim3((unsigned)((tot + 255) / 256)), 256, 0, stream>>>(w, o, Co, K, Kpad, tot);
  };
  auto conv = [&](const float* X, const unsigned short* Wp, const float* bias, float* Y,
                  int B, int Ci, int Hi, int Wi, int Co, int Ho, int Wo,
                  int KH, int KW, int s, int p, int ld, int Kpad, int relu) {
    int Ntot = B * Ho * Wo;
    int Mpad = ((Co + MT - 1) / MT) * MT;
    dim3 grid((Ntot + NT - 1) / NT, Mpad / MT);
    k_conv_gemm<<<grid, 256, 0, stream>>>(X, Wp, bias, Y, B, Ci, Hi, Wi, Co, Ho, Wo,
                                          KH, KW, s, p, ld, Kpad, relu);
  };
  auto bn = [&](float* buf, const float* g, const float* b, int B, int C, int HW, int relu) {
    k_bn_stats<<<C, 256, 0, stream>>>(buf, mean, istd, B, C, HW);
    long long tot = (long long)B * C * HW;
    k_bn_map<<<dim3((unsigned)((tot + 255) / 256)), 256, 0, stream>>>(
        buf, buf, mean, istd, g, b, C, HW, relu, tot);
  };
  auto resblock = [&](float* xbuf, float* t1, float* t2, int base,
                      unsigned short* pw3, unsigned short* pw1) {
    const float* b3  = (const float*)d_in[base + 1];
    const float* g3  = (const float*)d_in[base + 2];
    const float* bb3 = (const float*)d_in[base + 3];
    const float* b1  = (const float*)d_in[base + 5];
    const float* g1  = (const float*)d_in[base + 6];
    const float* bb1 = (const float*)d_in[base + 7];
    conv(xbuf, pw3, b3, t1, 32, 256, 64, 32, 256, 64, 32, 3, 3, 1, 1, 1, 2304, 1);
    bn(t1, g3, bb3, 32, 256, 64 * 32, 0);
    conv(t1, pw1, b1, t2, 32, 256, 64, 32, 256, 64, 32, 1, 1, 1, 0, 1, 256, 1);
    bn(t2, g1, bb1, 32, 256, 64 * 32, 0);
    k_add<<<dim3((16777216 + 255) / 256), 256, 0, stream>>>(xbuf, t2, 16777216LL);
  };

  // ---- 0. pack all weights to bf16 GEMM layout ----
  pack(enc_w1, p_enc1, 256, 48, 256, 64);
  pack(enc_w2, p_enc2, 256, 4096, 256, 4096);
  for (int r = 0; r < 4; ++r) {
    pack((const float*)d_in[rb_base[r] + 0], p_w3[r], 256, 2304, 256, 2304);
    pack((const float*)d_in[rb_base[r] + 4], p_w1[r], 256, 256, 256, 256);
  }
  pack(dec_w1, p_dec1, 256, 4096, 256, 4096);
  pack(dec_w2, p_dec2, 3, 4096, 64, 4096);
  pack(emb,    p_emb,  512, 256, 512, 256);
  k_emb_norm<<<2, 256, 0, stream>>>(emb, en, 512, 256);

  // ---- encoder ----
  conv(x, p_enc1, enc_b1, big0, 32, 3, 256, 128, 256, 128, 64, 4, 4, 2, 1, 1, 64, 0);
  bn(big0, bn1_g, bn1_b, 32, 256, 128 * 64, 1);
  conv(big0, p_enc2, enc_b2, midA, 32, 256, 128, 64, 256, 64, 32, 4, 4, 2, 1, 1, 4096, 0);
  resblock(midA, midB, midC, rb_base[0], p_w3[0], p_w1[0]);
  resblock(midA, midB, midC, rb_base[1], p_w3[1], p_w1[1]);   // midA = z_e_x
  k_copy<<<dim3((16777216 + 255) / 256), 256, 0, stream>>>(o_ze, midA, 16777216LL);

  // ---- vector quantization ----
  k_vq_argmin<<<dim3(65536 / 64), 128, 0, stream>>>(midA, p_emb, en, idx,
                                                    65536, 256, 64, 32, 256, 512);
  k_vq_gather<<<dim3((16777216 + 255) / 256), 256, 0, stream>>>(
      emb, idx, midB, o_zq, 256, 64, 32, 16777216LL);          // midB = z_q_x

  // ---- decoder ----
  resblock(midB, midA, midC, rb_base[2], p_w3[2], p_w1[2]);
  resblock(midB, midA, midC, rb_base[3], p_w3[3], p_w1[3]);
  // convT1: k4 s2 p1  ==  conv(stride1, pad2, lhs_dil2); input relu fused
  conv(midB, p_dec1, dec_b1, big0, 32, 256, 64, 32, 256, 128, 64, 4, 4, 1, 2, 2, 4096, 1);
  bn(big0, dbn_g, dbn_b, 32, 256, 128 * 64, 1);
  // convT2 -> x_tilde (raw), then tanh in place
  conv(big0, p_dec2, dec_b2, o_xt, 32, 256, 128, 64, 3, 256, 128, 4, 4, 1, 2, 2, 4096, 0);
  k_tanh<<<dim3((3145728 + 255) / 256), 256, 0, stream>>>(o_xt, 3145728LL);
}
